// GCNContext_26027501814021
// MI455X (gfx1250) — compile-verified, matrix operand order checked by reading, not run on hardware
//
#include <hip/hip_runtime.h>

// ---------------------------------------------------------------------------
// Types for CDNA5 WMMA
// ---------------------------------------------------------------------------
typedef __bf16 v16bf __attribute__((ext_vector_type(16)));
typedef float  v8f   __attribute__((ext_vector_type(8)));

#define N_NODES 50000
#define N_EDGES 800000
#define DIM     128
#define HID     256
#define BATCH   4096
#define SEQL    50

// ---------------------------------------------------------------------------
// f32 edge scatter-add (segment_sum of x[src] into agg[dst]); float4 gather.
// ---------------------------------------------------------------------------
__global__ void scatter_add_f32_kernel(const float* __restrict__ x,
                                       const int* __restrict__ src,
                                       const int* __restrict__ dst,
                                       float* __restrict__ agg,
                                       int nEdges, int feat) {
    int chunk = feat >> 2;
    long long tid = (long long)blockIdx.x * blockDim.x + threadIdx.x;
    long long total = (long long)nEdges * chunk;
    if (tid >= total) return;
    int e = (int)(tid / chunk);
    int c = (int)(tid % chunk);
    int s = src[e], d = dst[e];
    const float4 v = ((const float4*)(x + (size_t)s * feat))[c];
    float* out = agg + (size_t)d * feat + (size_t)c * 4;
    atomicAdd(out + 0, v.x);
    atomicAdd(out + 1, v.y);
    atomicAdd(out + 2, v.z);
    atomicAdd(out + 3, v.w);
}

// bf16 variant: x is a bf16 matrix with row stride `ld`, starting at column
// offset `colOff`; accumulation in f32 atomics.
__global__ void scatter_add_bf16_kernel(const __bf16* __restrict__ x,
                                        int ld, int colOff,
                                        const int* __restrict__ src,
                                        const int* __restrict__ dst,
                                        float* __restrict__ agg,
                                        int nEdges, int feat) {
    int chunk = feat >> 2;
    long long tid = (long long)blockIdx.x * blockDim.x + threadIdx.x;
    long long total = (long long)nEdges * chunk;
    if (tid >= total) return;
    int e = (int)(tid / chunk);
    int c = (int)(tid % chunk);
    int s = src[e], d = dst[e];
    const __bf16* row = x + (size_t)s * ld + colOff + (size_t)c * 4;
    float* out = agg + (size_t)d * feat + (size_t)c * 4;
    atomicAdd(out + 0, (float)row[0]);
    atomicAdd(out + 1, (float)row[1]);
    atomicAdd(out + 2, (float)row[2]);
    atomicAdd(out + 3, (float)row[3]);
}

__global__ void degree_kernel(const int* __restrict__ dst,
                              float* __restrict__ cnt, int nEdges) {
    int e = blockIdx.x * blockDim.x + threadIdx.x;
    if (e >= nEdges) return;
    atomicAdd(&cnt[dst[e]], 1.0f);
}

// ---------------------------------------------------------------------------
// f32 -> bf16 conversion with optional per-node mean divide; writes into a
// bf16 matrix with row stride ldout at column offset colOff.
// ---------------------------------------------------------------------------
__global__ void cvt_bf16_kernel(const float* __restrict__ in,
                                const float* __restrict__ cnt,   // null = no divide
                                __bf16* __restrict__ out,
                                int ldout, int colOff,
                                int nodes, int feat) {
    long long tid = (long long)blockIdx.x * blockDim.x + threadIdx.x;
    long long total = (long long)nodes * feat;
    if (tid >= total) return;
    int node = (int)(tid / feat);
    int f = (int)(tid % feat);
    float v = in[tid];
    if (cnt) v = v / fmaxf(cnt[node], 1.0f);
    out[(size_t)node * ldout + colOff + f] = (__bf16)v;
}

// Build concatenated transposed weight: out[n][k] for k in [0,2*Kh):
//   k <  Kh : Wl[k][n]        (Wl is [Kh][N] row-major f32)
//   k >= Kh : Wr[k-Kh][n]
__global__ void wcat_t_kernel(const float* __restrict__ Wl,
                              const float* __restrict__ Wr,
                              __bf16* __restrict__ out, int Kh, int N) {
    int K2 = 2 * Kh;
    long long tid = (long long)blockIdx.x * blockDim.x + threadIdx.x;
    long long total = (long long)N * K2;
    if (tid >= total) return;
    int n = (int)(tid / K2);
    int k = (int)(tid % K2);
    float v = (k < Kh) ? Wl[(size_t)k * N + n] : Wr[(size_t)(k - Kh) * N + n];
    out[tid] = (__bf16)v;
}

// Plain transpose: out[n][k] = W[k][n]  (W is [K][N] f32)
__global__ void wt_kernel(const float* __restrict__ W,
                          __bf16* __restrict__ out, int K, int N) {
    long long tid = (long long)blockIdx.x * blockDim.x + threadIdx.x;
    long long total = (long long)N * K;
    if (tid >= total) return;
    int n = (int)(tid / K);
    int k = (int)(tid % K);
    out[tid] = (__bf16)W[(size_t)k * N + n];
}

// ---------------------------------------------------------------------------
// bf16 WMMA GEMM with async global->LDS double-buffered staging.
//   C = act( A[M,K](bf16) @ Bt[N,K]^T(bf16) + bias + res )
// Output either f32 (Cf) or bf16 (Cb) with row stride ldc.
// 256 threads = 8 wave32; block tile 128x128; K-step 32; wave grid 4(M)x2(N);
// each wave: 2x4 v_wmma_f32_16x16x32_bf16 accumulators.
// N multiple of 128; K multiple of 32; M arbitrary (clamped loads, guarded
// stores). res (if set) is [M][N] f32 with ldc == N.
// ---------------------------------------------------------------------------
__launch_bounds__(256)
__global__ void gemm_bf16_wmma(const __bf16* __restrict__ A,
                               const __bf16* __restrict__ Bt,
                               const float* __restrict__ bias,
                               const float* __restrict__ res,
                               float* __restrict__ Cf,
                               __bf16* __restrict__ Cb,
                               int M, int N, int K, int ldc, int relu) {
    __shared__ __attribute__((aligned(16))) __bf16 As[2][128 * 32];  // [row][k]
    __shared__ __attribute__((aligned(16))) __bf16 Bs[2][128 * 32];  // [n][k]

    const int tid  = threadIdx.x;
    const int lane = tid & 31;
    const int wave = tid >> 5;
    const int wm   = wave >> 1;      // 0..3  -> 32-row slab
    const int wn   = wave & 1;       // 0..1  -> 64-col slab
    const int l15  = lane & 15;
    const int lhi  = lane >> 4;      // 0 or 1

    const int mBlock = blockIdx.x * 128;
    const int nBlock = blockIdx.y * 128;

    v8f acc[2][4];
    for (int i = 0; i < 2; i++)
        for (int j = 0; j < 4; j++)
            acc[i][j] = (v8f)(0.0f);

    // Issue async copies of one 128x32 bf16 tile pair into LDS buffer `buf`.
    // Each thread moves 2x16B of A and 2x16B of B (4 async B128 ops).
    auto issue_tiles = [&](int buf, int k0) {
        for (int i = 0; i < 2; i++) {
            int q  = tid + i * 256;          // 0..511
            int r  = q >> 2;                 // row 0..127
            int ch = (q & 3) << 3;           // bf16 offset 0,8,16,24 (16B chunks)
            int gm = mBlock + r;
            if (gm > M - 1) gm = M - 1;      // clamp tail rows (stores guarded)
            const __bf16* ga = A + (size_t)gm * K + k0 + ch;
            unsigned la = (unsigned)(size_t)(&As[buf][r * 32 + ch]);
            asm volatile("global_load_async_to_lds_b128 %0, %1, off"
                         :: "v"(la), "v"(ga) : "memory");
        }
        for (int i = 0; i < 2; i++) {
            int q  = tid + i * 256;
            int c  = q >> 2;                 // col 0..127
            int ch = (q & 3) << 3;
            const __bf16* gb = Bt + (size_t)(nBlock + c) * K + k0 + ch;
            unsigned lb = (unsigned)(size_t)(&Bs[buf][c * 32 + ch]);
            asm volatile("global_load_async_to_lds_b128 %0, %1, off"
                         :: "v"(lb), "v"(gb) : "memory");
        }
    };

    const int nsteps = K >> 5;
    issue_tiles(0, 0);

    for (int s = 0; s < nsteps; s++) {
        const int cur = s & 1;
        // Wait for this wave's outstanding async tile loads, then rendezvous.
        asm volatile("s_wait_asynccnt 0x0" ::: "memory");
        __syncthreads();
        // Prefetch next tile pair into the other buffer while computing.
        if (s + 1 < nsteps) issue_tiles(cur ^ 1, (s + 1) * 32);

        // ---- load fragments (ISA 16-bit 16x32 A layout; B symmetric) ----
        union Frag { v16bf v; unsigned u[8]; };
        Frag af[2], bf[4];
        const unsigned* Au = (const unsigned*)As[cur];
        const unsigned* Bu = (const unsigned*)Bs[cur];
        for (int t = 0; t < 2; t++) {
            int row  = wm * 32 + t * 16 + l15;
            int base = row * 16 + 4 * lhi;            // 16 uints per 32-bf16 row
            af[t].u[0] = Au[base + 0];  af[t].u[1] = Au[base + 1];
            af[t].u[2] = Au[base + 2];  af[t].u[3] = Au[base + 3];
            af[t].u[4] = Au[base + 8];  af[t].u[5] = Au[base + 9];
            af[t].u[6] = Au[base + 10]; af[t].u[7] = Au[base + 11];
        }
        for (int t = 0; t < 4; t++) {
            int col  = wn * 64 + t * 16 + l15;
            int base = col * 16 + 4 * lhi;
            bf[t].u[0] = Bu[base + 0];  bf[t].u[1] = Bu[base + 1];
            bf[t].u[2] = Bu[base + 2];  bf[t].u[3] = Bu[base + 3];
            bf[t].u[4] = Bu[base + 8];  bf[t].u[5] = Bu[base + 9];
            bf[t].u[6] = Bu[base + 10]; bf[t].u[7] = Bu[base + 11];
        }
        // ---- 8 WMMAs per wave per K-step ----
        for (int tm = 0; tm < 2; tm++)
            for (int tn = 0; tn < 4; tn++)
                acc[tm][tn] = __builtin_amdgcn_wmma_f32_16x16x32_bf16(
                    false, af[tm].v, false, bf[tn].v,
                    (short)0, acc[tm][tn], false, false);
    }

    // ---- epilogue: bias + residual + optional ReLU; f32 or bf16 store ----
    for (int tm = 0; tm < 2; tm++) {
        for (int tn = 0; tn < 4; tn++) {
            int col = nBlock + wn * 64 + tn * 16 + l15;
            float bval = bias ? bias[col] : 0.0f;
            for (int j = 0; j < 8; j++) {
                int row = mBlock + wm * 32 + tm * 16 + j + 8 * lhi;
                if (row < M) {
                    float v = acc[tm][tn][j] + bval;
                    size_t off = (size_t)row * ldc + col;
                    if (res) v += res[(size_t)row * N + col];
                    if (relu) v = fmaxf(v, 0.0f);
                    if (Cf) Cf[off] = v;
                    else    Cb[off] = (__bf16)v;
                }
            }
        }
    }
}

// ---------------------------------------------------------------------------
// Gather + sum over token dim; emd[b][0:128]=sum sent, [128:256]=sum ctx
// ---------------------------------------------------------------------------
__global__ void gather_sum_kernel(const float* __restrict__ x,
                                  const int* __restrict__ sent,
                                  const int* __restrict__ ctx,
                                  float* __restrict__ emd) {
    int b = blockIdx.x;       // 0..4095
    int d = threadIdx.x;      // 0..127
    float s = 0.0f, c = 0.0f;
    const int* sp = sent + b * SEQL;
    const int* cp = ctx + b * SEQL;
    for (int l = 0; l < SEQL; l++) {
        s += x[(size_t)sp[l] * DIM + d];
        c += x[(size_t)cp[l] * DIM + d];
    }
    emd[(size_t)b * 256 + d]       = s;
    emd[(size_t)b * 256 + 128 + d] = c;
}

// ---------------------------------------------------------------------------
// BatchNorm1d with batch statistics; one block per feature; bf16 output.
// ---------------------------------------------------------------------------
__global__ void batchnorm_kernel(const float* __restrict__ emd,
                                 const float* __restrict__ gamma,
                                 const float* __restrict__ beta,
                                 __bf16* __restrict__ out) {
    int f = blockIdx.x;       // 0..255
    __shared__ float red[256];
    float s = 0.0f;
    for (int b = threadIdx.x; b < BATCH; b += 256)
        s += emd[(size_t)b * 256 + f];
    red[threadIdx.x] = s; __syncthreads();
    for (int st = 128; st > 0; st >>= 1) {
        if (threadIdx.x < st) red[threadIdx.x] += red[threadIdx.x + st];
        __syncthreads();
    }
    float mu = red[0] * (1.0f / BATCH);
    __syncthreads();
    float v = 0.0f;
    for (int b = threadIdx.x; b < BATCH; b += 256) {
        float d = emd[(size_t)b * 256 + f] - mu;
        v += d * d;
    }
    red[threadIdx.x] = v; __syncthreads();
    for (int st = 128; st > 0; st >>= 1) {
        if (threadIdx.x < st) red[threadIdx.x] += red[threadIdx.x + st];
        __syncthreads();
    }
    float var = red[0] * (1.0f / BATCH);
    float inv = rsqrtf(var + 1e-5f) * gamma[f];
    float be  = beta[f];
    for (int b = threadIdx.x; b < BATCH; b += 256)
        out[(size_t)b * 256 + f] =
            (__bf16)((emd[(size_t)b * 256 + f] - mu) * inv + be);
}

// ---------------------------------------------------------------------------
// Tiny final layer: [4096,512] @ [512,2] + b
// ---------------------------------------------------------------------------
__global__ void fc2_kernel(const float* __restrict__ h1,
                           const float* __restrict__ w,
                           const float* __restrict__ b,
                           float* __restrict__ out) {
    int tid = blockIdx.x * blockDim.x + threadIdx.x;   // 0..8191
    if (tid >= BATCH * 2) return;
    int row = tid >> 1, col = tid & 1;
    float acc = b[col];
    const float* hr = h1 + (size_t)row * 512;
    for (int k = 0; k < 512; k++)
        acc += hr[k] * w[k * 2 + col];
    out[tid] = acc;
}

// ---------------------------------------------------------------------------
// Host launcher
// ---------------------------------------------------------------------------
static inline unsigned blocks_for(long long total, int tpb) {
    return (unsigned)((total + tpb - 1) / tpb);
}

extern "C" void kernel_launch(void* const* d_in, const int* in_sizes, int n_in,
                              void* d_out, int out_size, void* d_ws, size_t ws_size,
                              hipStream_t stream) {
    const int*   sentence = (const int*)d_in[0];
    const int*   context  = (const int*)d_in[1];
    const int*   edge     = (const int*)d_in[2];
    const float* emb      = (const float*)d_in[3];
    const float* Wl1      = (const float*)d_in[4];
    const float* bl1      = (const float*)d_in[5];
    const float* Wr1      = (const float*)d_in[6];
    const float* Wl2      = (const float*)d_in[7];
    const float* bl2      = (const float*)d_in[8];
    const float* Wr2      = (const float*)d_in[9];
    const float* gamma    = (const float*)d_in[10];
    const float* beta     = (const float*)d_in[11];
    const float* fc1_w    = (const float*)d_in[12];
    const float* fc1_b    = (const float*)d_in[13];
    const float* fc2_w    = (const float*)d_in[14];
    const float* fc2_b    = (const float*)d_in[15];
    float* out = (float*)d_out;

    const int* src = edge;
    const int* dst = edge + N_EDGES;

    // Workspace carve-up
    char* p = (char*)d_ws;
    float* cnt = (float*)p;    p += (size_t)50016 * 4;
    float* agg = (float*)p;    p += (size_t)N_NODES * HID * 4;    // 51.2 MB (layer2 max)
    __bf16* A1b = (__bf16*)p;  p += (size_t)N_NODES * 256 * 2;    // [mean1|emb] bf16
    __bf16* A2b = (__bf16*)p;  p += (size_t)N_NODES * 512 * 2;    // [mean2|x1] bf16
    float* x2  = (float*)p;    p += (size_t)N_NODES * DIM * 4;
    float* emd = (float*)p;    p += (size_t)BATCH * 256 * 4;
    __bf16* emdnb = (__bf16*)p; p += (size_t)BATCH * 256 * 2;
    float* h1  = (float*)p;    p += (size_t)BATCH * 512 * 4;
    __bf16* wc1 = (__bf16*)p;  p += (size_t)256 * 256 * 2;        // [Wl1;Wr1]^T
    __bf16* wc2 = (__bf16*)p;  p += (size_t)128 * 512 * 2;        // [Wl2;Wr2]^T
    __bf16* fct = (__bf16*)p;  p += (size_t)512 * 256 * 2;        // fc1_w^T

    // ---- weight prep (tiny, once per call) ----
    wcat_t_kernel<<<blocks_for(256LL * 256, 256), 256, 0, stream>>>(Wl1, Wr1, wc1, 128, 256);
    wcat_t_kernel<<<blocks_for(128LL * 512, 256), 256, 0, stream>>>(Wl2, Wr2, wc2, 256, 128);
    wt_kernel<<<blocks_for(512LL * 256, 256), 256, 0, stream>>>(fc1_w, fct, 256, 512);

    // ---------------- Layer 1: SAGEConv(emb) ----------------
    hipMemsetAsync(cnt, 0, sizeof(float) * N_NODES, stream);
    hipMemsetAsync(agg, 0, sizeof(float) * (size_t)N_NODES * DIM, stream);
    degree_kernel<<<blocks_for(N_EDGES, 256), 256, 0, stream>>>(dst, cnt, N_EDGES);
    scatter_add_f32_kernel<<<blocks_for((long long)N_EDGES * (DIM / 4), 256), 256, 0, stream>>>(
        emb, src, dst, agg, N_EDGES, DIM);
    // A1b = [ mean1 | emb ] in bf16
    cvt_bf16_kernel<<<blocks_for((long long)N_NODES * DIM, 256), 256, 0, stream>>>(
        agg, cnt, A1b, 256, 0, N_NODES, DIM);
    cvt_bf16_kernel<<<blocks_for((long long)N_NODES * DIM, 256), 256, 0, stream>>>(
        emb, nullptr, A1b, 256, 128, N_NODES, DIM);
    // x1 = relu(A1b @ wc1 + bl1) -> bf16 into A2b columns [256,512)
    {
        dim3 grid((N_NODES + 127) / 128, HID / 128);
        gemm_bf16_wmma<<<grid, 256, 0, stream>>>(A1b, wc1, bl1, nullptr,
                                                 nullptr, A2b + 256,
                                                 N_NODES, HID, 256, 512, 1);
    }

    // ---------------- Layer 2: SAGEConv(x1) + residual emb ----------------
    hipMemsetAsync(agg, 0, sizeof(float) * (size_t)N_NODES * HID, stream);
    scatter_add_bf16_kernel<<<blocks_for((long long)N_EDGES * (HID / 4), 256), 256, 0, stream>>>(
        A2b, 512, 256, src, dst, agg, N_EDGES, HID);
    // A2b columns [0,256) = mean2 bf16
    cvt_bf16_kernel<<<blocks_for((long long)N_NODES * HID, 256), 256, 0, stream>>>(
        agg, cnt, A2b, 512, 0, N_NODES, HID);
    // x2 = A2b @ wc2 + bl2 + emb  (f32)
    {
        dim3 grid((N_NODES + 127) / 128, DIM / 128);
        gemm_bf16_wmma<<<grid, 256, 0, stream>>>(A2b, wc2, bl2, emb,
                                                 x2, nullptr,
                                                 N_NODES, DIM, 512, DIM, 0);
    }

    // ---------------- Readout: gather+sum, BN, MLP ----------------
    gather_sum_kernel<<<BATCH, DIM, 0, stream>>>(x2, sentence, context, emd);
    batchnorm_kernel<<<256, 256, 0, stream>>>(emd, gamma, beta, emdnb);
    {
        dim3 grid(BATCH / 128, 512 / 128);
        gemm_bf16_wmma<<<grid, 256, 0, stream>>>(emdnb, fct, fc1_b, nullptr,
                                                 h1, nullptr,
                                                 BATCH, 512, 256, 512, 1);
    }
    fc2_kernel<<<blocks_for(BATCH * 2, 256), 256, 0, stream>>>(h1, fc2_w, fc2_b, out);
}
